// TripletLoss_1022202217030
// MI455X (gfx1250) — compile-verified, matrix operand order checked by reading, not run on hardware
//
#include <hip/hip_runtime.h>
#include <stdint.h>
#include <math.h>

#define MARGIN 0.5f

// ---------------- CDNA5 feature detection ----------------
#if __has_include(<hip/amd_detail/amd_gfx1250_TDM.h>)
#define TDM_SIX_ARGS 1
#endif

#if __has_builtin(__builtin_amdgcn_tensor_load_to_lds)
#define HAVE_TDM 1
#else
#define HAVE_TDM 0
#endif

#if __has_builtin(__builtin_amdgcn_s_wait_tensorcnt)
#define WAIT_TENSOR(n) __builtin_amdgcn_s_wait_tensorcnt(n)
#else
#define WAIT_TENSOR(n) do {} while (0)
#endif

typedef _Float16 v16h __attribute__((ext_vector_type(16)));
typedef float    v8f  __attribute__((ext_vector_type(8)));
typedef unsigned tdm_u32x4 __attribute__((ext_vector_type(4)));
typedef int      tdm_i32x8 __attribute__((ext_vector_type(8)));
typedef int      tdm_i32x4 __attribute__((ext_vector_type(4)));

// =========================================================
// Kernel 1: matched[j] (argmax index) and has_match[j] via
// exact one-hot f16 WMMA: matched = iota^T * gt, cnt = 1^T * gt.
// One wave per 16 columns.  D-layout: VGPR0, lanes 0..15 -> N=lane.
// =========================================================
__global__ __launch_bounds__(32) void match_kernel(const float* __restrict__ gt,
                                                   float* __restrict__ matched_f,
                                                   float* __restrict__ hasm) {
  const int lane = threadIdx.x;
  const int j0 = blockIdx.x * 16;
  const int n  = lane & 15;
  const int hs = (lane >> 4) & 1;   // half-wave select

  v8f acc_idx = {};
  v8f acc_cnt = {};

  for (int i0 = 0; i0 < 1024; i0 += 32) {
    v16h A_iota, A_one, B;
#pragma unroll
    for (int h = 0; h < 16; ++h) {
      // 16-bit operand K mapping (ISA 7.12.2): VGPR v halves h=2v,2v+1,
      // lanes 0-15: K=h (+16 for h>=8's upper block), lanes 16-31: +8.
      const int K = h + 8 * hs + ((h & 8) ? 8 : 0);
      A_iota[h] = (_Float16)(float)(i0 + K);  // exact: <= 1023 < 2048
      A_one[h]  = (_Float16)1.0f;
      B[h]      = (_Float16)gt[(i0 + K) * 1024 + (j0 + n)];  // exact: 0.0 / 1.0
    }
    acc_idx = __builtin_amdgcn_wmma_f32_16x16x32_f16(false, A_iota, false, B,
                                                     (short)0, acc_idx, false, false);
    acc_cnt = __builtin_amdgcn_wmma_f32_16x16x32_f16(false, A_one, false, B,
                                                     (short)0, acc_cnt, false, false);
  }

  if (lane < 16) {
    matched_f[j0 + n] = acc_idx[0];  // all M rows identical; row M=0
    hasm[j0 + n]      = acc_cnt[0];  // column sum; !=0 means has_match
  }
}

// =========================================================
// Kernel 2: Ddt[i,j] = sum_k |det[i,k]-trk[j,k]|.
// 64x64 output tile per 256-thread block, k in 8 chunks of 32,
// TDM tensor_load_to_lds (double-buffered, TENSORcnt) stages
// 64x32 f32 tiles with pad(4 dw every 32 dw) -> LDS stride 36
// for conflict-free b128 reads.
// =========================================================
__device__ inline void tdm_tile_load(unsigned lds_off, const float* gptr) {
#if HAVE_TDM
  tdm_u32x4 g0;
  const unsigned long long ga = (unsigned long long)(uintptr_t)gptr;
  g0[0] = 1u;                                   // count=1, user descriptor
  g0[1] = lds_off;                              // LDS byte address
  g0[2] = (unsigned)ga;                         // global_addr[31:0]
  g0[3] = ((unsigned)(ga >> 32) & 0x01FFFFFFu)  // global_addr[56:32]
          | (2u << 30);                         // type=2 ("image")
  tdm_i32x8 g1;
  g1[0] = (int)((2u << 16)    // data_size = 4B
              | (1u << 20)    // pad_enable
              | (4u << 22)    // pad_interval code 4 = every 32 DWORDs
              | (3u << 25));  // pad_amount  code 3 = 4 DWORDs
  g1[1] = (int)(256u << 16);  // tensor_dim0 = 256
  g1[2] = (int)(2048u << 16); // tensor_dim1 = 2048 (generous, no clipping)
  g1[3] = (int)(32u << 16);   // tile_dim0 = 32 elements
  g1[4] = 64;                 // tile_dim1 = 64 rows, tile_dim2 = 0
  g1[5] = 256;                // tensor_dim0_stride = 256 elements
  g1[6] = 0;
  g1[7] = 0;
  tdm_i32x4 z4 = {0, 0, 0, 0};
#if defined(TDM_SIX_ARGS)
  tdm_i32x8 z8 = {0, 0, 0, 0, 0, 0, 0, 0};
  __builtin_amdgcn_tensor_load_to_lds(g0, g1, z4, z4, z8, 0);
#else
  __builtin_amdgcn_tensor_load_to_lds(g0, g1, z4, z4, 0);
#endif
#else
  (void)lds_off; (void)gptr;
#endif
}

__device__ inline void l1_tile_accum(const float* __restrict__ dB,
                                     const float* __restrict__ tB,
                                     int tx, int ty, float acc[4][4]) {
#pragma unroll
  for (int kk = 0; kk < 32; kk += 4) {
    float4 dv[4], tv[4];
#pragma unroll
    for (int s = 0; s < 4; ++s)
      dv[s] = *(const float4*)(dB + (ty + 16 * s) * 36 + kk);
#pragma unroll
    for (int t = 0; t < 4; ++t)
      tv[t] = *(const float4*)(tB + (tx + 16 * t) * 36 + kk);
#pragma unroll
    for (int s = 0; s < 4; ++s)
#pragma unroll
      for (int t = 0; t < 4; ++t)
        acc[s][t] += fabsf(dv[s].x - tv[t].x) + fabsf(dv[s].y - tv[t].y) +
                     fabsf(dv[s].z - tv[t].z) + fabsf(dv[s].w - tv[t].w);
  }
}

__global__ __launch_bounds__(256) void ddt_kernel(const float* __restrict__ emb,
                                                  float* __restrict__ Ddt) {
  extern __shared__ float smem[];  // 2 bufs x (det 64x36 + trk 64x36) = 36864 B
  const int tid = threadIdx.x;
  const int tx = tid & 15, ty = tid >> 4;
  const int bi = blockIdx.y, bj = blockIdx.x;
  const int r0 = bi * 64;  // det rows
  const int c0 = bj * 64;  // trk rows (= Ddt cols)
  const float* det = emb + (size_t)r0 * 256;
  const float* trk = emb + (size_t)(1024 + c0) * 256;

  float acc[4][4] = {{0.f, 0.f, 0.f, 0.f}, {0.f, 0.f, 0.f, 0.f},
                     {0.f, 0.f, 0.f, 0.f}, {0.f, 0.f, 0.f, 0.f}};

#if HAVE_TDM
  const unsigned ldsbase = (unsigned)(uintptr_t)smem;  // flat low 32 bits = LDS offset
  if (tid < 32) {  // wave 0 drives the Tensor Data Mover
    tdm_tile_load(ldsbase + 0u,    det);
    tdm_tile_load(ldsbase + 9216u, trk);
  }
  for (int c = 0; c < 8; ++c) {
    if (tid < 32) {
      if (c + 1 < 8) {
        const unsigned b = ((unsigned)(c + 1) & 1u) * 18432u;
        tdm_tile_load(ldsbase + b,         det + (c + 1) * 32);
        tdm_tile_load(ldsbase + b + 9216u, trk + (c + 1) * 32);
      }
      if (c + 1 < 8) WAIT_TENSOR(2);  // chunk c's two TDM ops complete (in-order)
      else           WAIT_TENSOR(0);
    }
    __syncthreads();
    const float* dB = smem + (c & 1) * 4608;
    const float* tB = dB + 2304;
    l1_tile_accum(dB, tB, tx, ty, acc);
    __syncthreads();  // buffer (c&1) safe to overwrite at iteration c+1
  }
#else
  for (int c = 0; c < 8; ++c) {
    __syncthreads();
    for (int idx = tid; idx < 64 * 32; idx += 256) {
      const int r = idx >> 5, k = idx & 31;
      smem[r * 36 + k]        = det[r * 256 + c * 32 + k];
      smem[2304 + r * 36 + k] = trk[r * 256 + c * 32 + k];
    }
    __syncthreads();
    l1_tile_accum(smem, smem + 2304, tx, ty, acc);
  }
#endif

#pragma unroll
  for (int s = 0; s < 4; ++s)
#pragma unroll
    for (int t = 0; t < 4; ++t)
      Ddt[(size_t)(r0 + ty + 16 * s) * 1024 + (c0 + tx + 16 * t)] = acc[s][t];
}

// =========================================================
// Kernel 3: per-column j: t1, t2 (col-min excluding matched row),
// zero_val (col-min + margin), t3 (row-min of gathered row, j2!=j),
// contrib = relu(t1 - t2 - t3 + margin).
// =========================================================
__global__ __launch_bounds__(256) void perj_kernel(const float* __restrict__ emb,
                                                   const float* __restrict__ Ddt,
                                                   const float* __restrict__ matched_f,
                                                   float* __restrict__ contrib,
                                                   float* __restrict__ zeroval) {
  const int j = blockIdx.x * 256 + threadIdx.x;
  if (j >= 1024) return;

  const float mf = matched_f[j];
  int mi = (int)(mf + 0.5f);
  mi = mi < 0 ? 0 : (mi > 1023 ? 1023 : mi);

  // t1 = sum_k |trk[j,k] - (float)matched[j]|  (reference subtracts the index!)
  const float* trk = emb + (size_t)(1024 + j) * 256;
  float t1 = 0.f;
  for (int k = 0; k < 256; ++k) t1 += fabsf(trk[k] - mf);

  float cmin = 3.402823466e38f, cex = 3.402823466e38f;
  for (int i = 0; i < 1024; ++i) {
    const float v = Ddt[(size_t)i * 1024 + j];
    cmin = fminf(cmin, v);
    if (i != mi) cex = fminf(cex, v);
  }

  float t3 = 3.402823466e38f;
  const float* row = Ddt + (size_t)mi * 1024;
  for (int j2 = 0; j2 < 1024; ++j2)
    if (j2 != j) t3 = fminf(t3, row[j2]);

  contrib[j] = fmaxf(t1 - cex - t3 + MARGIN, 0.f);
  zeroval[j] = cmin + MARGIN;
}

// =========================================================
// Kernel 4: the sequential scan (no-match columns RESET the loss).
// =========================================================
__global__ void final_kernel(const float* __restrict__ hasm,
                             const float* __restrict__ contrib,
                             const float* __restrict__ zeroval,
                             float* __restrict__ out) {
  if (threadIdx.x == 0 && blockIdx.x == 0) {
    float loss = 0.f;
    for (int j = 0; j < 1024; ++j)
      loss = (hasm[j] != 0.f) ? (loss + contrib[j]) : zeroval[j];
    out[0] = loss;
  }
}

// =========================================================
extern "C" void kernel_launch(void* const* d_in, const int* in_sizes, int n_in,
                              void* d_out, int out_size, void* d_ws, size_t ws_size,
                              hipStream_t stream) {
  (void)in_sizes; (void)n_in; (void)out_size; (void)ws_size;
  const float* emb = (const float*)d_in[0];   // (2048, 256) f32
  const float* gt  = (const float*)d_in[1];   // (1024, 1024) f32

  float* W         = (float*)d_ws;
  float* Ddt       = W;                       // 1024*1024 f32 = 4 MB
  float* matched_f = W + (1u << 20);
  float* hasm      = matched_f + 1024;
  float* contrib   = hasm + 1024;
  float* zeroval   = contrib + 1024;

  match_kernel<<<64, 32, 0, stream>>>(gt, matched_f, hasm);
  ddt_kernel<<<dim3(16, 16, 1), dim3(256, 1, 1), 36864, stream>>>(emb, Ddt);
  perj_kernel<<<4, 256, 0, stream>>>(emb, Ddt, matched_f, contrib, zeroval);
  final_kernel<<<1, 1, 0, stream>>>(hasm, contrib, zeroval, (float*)d_out);
}
